// PoseConvLSTM_90366111908538
// MI455X (gfx1250) — matrix operand order repaired and checked by reading, not compiled, not run on hardware
//
#include <hip/hip_runtime.h>
#include <hip/hip_bf16.h>

typedef __attribute__((ext_vector_type(16))) _Float16 v16h;
typedef __attribute__((ext_vector_type(8)))  _Float16 v8h;
typedef __attribute__((ext_vector_type(8)))  float    v8f;

union V16 { v16h v; v8h h[2]; };

__device__ __forceinline__ float sigf(float x) { return 1.0f / (1.0f + __expf(-x)); }

// ---------------------------------------------------------------------------
// Convert f32 weights to K-padded f16 rows.
// wh1: [128][320] (K=315 padded to 320), wh2: [256][864] (864 = 27*32)
// ---------------------------------------------------------------------------
__global__ __launch_bounds__(256) void convert_weights(
    const float* __restrict__ w1, const float* __restrict__ w2,
    _Float16* __restrict__ wh1, _Float16* __restrict__ wh2) {
  int i = blockIdx.x * 256 + threadIdx.x;
  const int N1 = 128 * 320;
  const int N2 = 256 * 864;
  if (i < N1) {
    int m = i / 320, k = i - m * 320;
    wh1[i] = (k < 315) ? (_Float16)w1[m * 315 + k] : (_Float16)0.0f;
  } else if (i < N1 + N2) {
    int j = i - N1;
    wh2[j] = (_Float16)w2[j];
  }
}

__global__ __launch_bounds__(256) void zero_kernel(float* p, int n) {
  int i = blockIdx.x * 256 + threadIdx.x;
  if (i < n) p[i] = 0.0f;
}

// ---------------------------------------------------------------------------
// One fused ConvLSTM timestep via implicit-GEMM WMMA.
// Block: 256 threads (8 waves), owns 16 contiguous pixels (one row segment)
// and the FULL M = 4*HC gate channels. Grid: 4096/16 = 256 blocks.
//
// Phase 1: stage halo tile S = concat(x,h)[CT][3][18] (SAME pad) -> LDS f16
// Phase 2: build full im2col panel Bt[16 pixels][KP] in LDS (once)
// Phase 3: barrier-free K-loop of v_wmma_f32_16x16x32_f16
// Phase 4: gates (sigmoid/tanh) fused through LDS, h/c update
// ---------------------------------------------------------------------------
template <int HC, int CIN>
__global__ __launch_bounds__(256) void convlstm_step(
    const float* __restrict__ x,     // [CIN][64][64]
    const float* __restrict__ hin,   // [HC][64][64]
    float* __restrict__ hout,        // [HC][64][64]
    float* __restrict__ cst,         // [HC][64][64], updated in place
    const _Float16* __restrict__ wh, // [4*HC][KP] f16, K-padded
    const float* __restrict__ bias)  // [4*HC]
{
  constexpr int CT = CIN + HC;             // input channels to the conv
  constexpr int KD = CT * 9;               // 315 or 864
  constexpr int KP = (KD + 31) / 32 * 32;  // 320 or 864
  constexpr int M   = 4 * HC;
  constexpr int MT  = M / 16;              // 8 or 16 M-tiles
  constexpr int TPW = MT / 8;              // tiles per wave: 1 or 2
  constexpr int SSTR = 20;                 // halo row stride (halves)

  __shared__ __align__(32) _Float16 S[CT * 3 * SSTR];  // halo tile
  __shared__ __align__(32) _Float16 Bt[16 * KP];       // im2col panel
  __shared__ float Zt[M * 16];                         // pre-gate z

  const int tid  = threadIdx.x;
  const int lane = tid & 31;
  const int wv   = tid >> 5;               // wave 0..7
  const int pbase = blockIdx.x * 16;
  const int ybase = pbase >> 6;            // fixed row for this block
  const int xbase = pbase & 63;            // 16-aligned column start

  // ---- Phase 1: halo staging (coalesced, converts f32 -> f16) ----
  for (int e = tid; e < CT * 3 * 18; e += 256) {
    int xi = e % 18;
    int t2 = e / 18;
    int yi = t2 % 3;
    int c  = t2 / 3;
    int gy = ybase + yi - 1;
    int gx = xbase + xi - 1;
    float v = 0.0f;
    if ((unsigned)gy < 64u && (unsigned)gx < 64u)
      v = (c < CIN) ? x[c * 4096 + gy * 64 + gx]
                    : hin[(c - CIN) * 4096 + gy * 64 + gx];
    S[(c * 3 + yi) * SSTR + xi] = (_Float16)v;
  }
  __syncthreads();

  // ---- Phase 2: im2col panel Bt[n][k] = S[(c*3+ky)*SSTR + n + kx] ----
  {
    int n = tid >> 4;                      // 16 threads per pixel row
    for (int k = (tid & 15); k < KD; k += 16) {
      int c = k / 9;
      int r = k - c * 9;
      int ky = r / 3, kx = r - ky * 3;
      Bt[n * KP + k] = S[(c * 3 + ky) * SSTR + n + kx];
    }
    if constexpr (KP > KD) {
      for (int k = KD + (tid & 15); k < KP; k += 16)
        Bt[n * KP + k] = (_Float16)0.0f;
    }
  }
  __syncthreads();

  // ---- Phase 3: barrier-free WMMA K-loop ----
  v8f acc[TPW];
  const v8f vz = {0.f, 0.f, 0.f, 0.f, 0.f, 0.f, 0.f, 0.f};
#pragma unroll
  for (int t = 0; t < TPW; ++t) acc[t] = vz;

  // Lane mappings used here (16-bit operands, contiguous k order shared by A and B):
  //  B (32x16): lanes 0-15 -> N=lane, K 0..15; lanes 16-31 -> N=lane-16, K 16..31
  //  A (16x32): lanes 0-15 -> K 0..15; lanes 16-31 -> K 16..31
  const int nB  = lane & 15;
  const int k0B = (lane >> 4) * 16;
  const int mA  = lane & 15;
  const int kbA = (lane >> 4) * 16;

  for (int kk = 0; kk < KP; kk += 32) {
    V16 b;
    b.v = *(const v16h*)&Bt[nB * KP + k0B + kk];
#pragma unroll
    for (int t = 0; t < TPW; ++t) {
      int mt = wv + t * 8;
      const _Float16* arow = wh + (mt * 16 + mA) * KP + kk;
      V16 a;
      a.h[0] = *(const v8h*)(arow + kbA);
      a.h[1] = *(const v8h*)(arow + kbA + 8);
      acc[t] = __builtin_amdgcn_wmma_f32_16x16x32_f16(
          false, a.v, false, b.v, (short)0, acc[t], false, false);
    }
  }

  // ---- C layout: VGPR v -> M = v + (lane>=16 ? 8 : 0), N = lane & 15 ----
  __syncthreads();
#pragma unroll
  for (int t = 0; t < TPW; ++t) {
    int mt = wv + t * 8;
    int n  = lane & 15;
    int mo = (lane >> 4) * 8;
#pragma unroll
    for (int v = 0; v < 8; ++v) {
      int mg = mt * 16 + v + mo;
      Zt[mg * 16 + n] = acc[t][v] + bias[mg];
    }
  }
  __syncthreads();

  // ---- Phase 4: gates. i,f,o,g = split(z,4) ----
  for (int idx = tid; idx < HC * 16; idx += 256) {
    int c = idx >> 4;
    int n = idx & 15;
    float zi = Zt[c * 16 + n];
    float zf = Zt[(HC + c) * 16 + n];
    float zo = Zt[(2 * HC + c) * 16 + n];
    float zg = Zt[(3 * HC + c) * 16 + n];
    int p = pbase + n;
    float cp = cst[c * 4096 + p];
    float cn = sigf(zf) * cp + sigf(zi) * tanhf(zg);
    float hn = sigf(zo) * tanhf(cn);
    cst[c * 4096 + p] = cn;
    hout[c * 4096 + p] = hn;
  }
}

// ---------------------------------------------------------------------------
// FC head: out[j] = fc_b[j] + dot(fc_w[j], ys2)   (6 blocks, 1024 threads)
// ---------------------------------------------------------------------------
__global__ __launch_bounds__(1024) void fc_kernel(
    const float* __restrict__ ys2,  // [262144]
    const float* __restrict__ fcw,  // [6][262144]
    const float* __restrict__ fcb,  // [6]
    float* __restrict__ out)        // [6] for this timestep
{
  __shared__ float red[1024];
  const int j = blockIdx.x;
  const float* wr = fcw + j * 262144;
  float s = 0.0f;
  for (int e = threadIdx.x * 4; e < 262144; e += 1024 * 4) {
    float4 a = *(const float4*)(ys2 + e);
    float4 w = *(const float4*)(wr + e);
    s += a.x * w.x + a.y * w.y + a.z * w.z + a.w * w.w;
  }
  red[threadIdx.x] = s;
  __syncthreads();
  for (int off = 512; off > 0; off >>= 1) {
    if (threadIdx.x < off) red[threadIdx.x] += red[threadIdx.x + off];
    __syncthreads();
  }
  if (threadIdx.x == 0) out[j] = red[0] + fcb[j];
}

// ---------------------------------------------------------------------------
extern "C" void kernel_launch(void* const* d_in, const int* in_sizes, int n_in,
                              void* d_out, int out_size, void* d_ws, size_t ws_size,
                              hipStream_t stream) {
  (void)in_sizes; (void)n_in; (void)out_size; (void)ws_size;
  const float* input = (const float*)d_in[0]; // [128][3][64][64]
  const float* w1    = (const float*)d_in[1]; // [128][35][3][3]
  const float* b1    = (const float*)d_in[2]; // [128]
  const float* w2    = (const float*)d_in[3]; // [256][96][3][3]
  const float* b2    = (const float*)d_in[4]; // [256]
  const float* fcw   = (const float*)d_in[5]; // [6][262144]
  const float* fcb   = (const float*)d_in[6]; // [6]
  float* out = (float*)d_out;

  // Workspace layout (256B aligned slices)
  char* ws = (char*)d_ws;
  size_t off = 0;
  auto alloc = [&](size_t bytes) -> char* {
    char* p = ws + off;
    off += (bytes + 255) & ~size_t(255);
    return p;
  };
  _Float16* wh1 = (_Float16*)alloc(128 * 320 * 2);
  _Float16* wh2 = (_Float16*)alloc(256 * 864 * 2);
  float* h1 = (float*)alloc(2 * 131072 * 4); // ping-pong [32][64][64]
  float* c1 = (float*)alloc(131072 * 4);
  float* h2 = (float*)alloc(2 * 262144 * 4); // ping-pong [64][64][64]
  float* c2 = (float*)alloc(262144 * 4);

  // Weights -> f16 padded rows
  {
    int tot = 128 * 320 + 256 * 864;
    convert_weights<<<(tot + 255) / 256, 256, 0, stream>>>(w1, w2, wh1, wh2);
  }
  // Zero initial states (only buf0 of h ping-pongs is read at t=0)
  zero_kernel<<<(131072 + 255) / 256, 256, 0, stream>>>(h1, 131072);
  zero_kernel<<<(131072 + 255) / 256, 256, 0, stream>>>(c1, 131072);
  zero_kernel<<<(262144 + 255) / 256, 256, 0, stream>>>(h2, 262144);
  zero_kernel<<<(262144 + 255) / 256, 256, 0, stream>>>(c2, 262144);

  for (int t = 0; t < 128; ++t) {
    const float* xt = input + (size_t)t * 3 * 4096;
    float* h1in  = h1 + (t & 1) * 131072;
    float* h1out = h1 + ((t + 1) & 1) * 131072;
    convlstm_step<32, 3><<<256, 256, 0, stream>>>(xt, h1in, h1out, c1, wh1, b1);

    float* h2in  = h2 + (t & 1) * 262144;
    float* h2out = h2 + ((t + 1) & 1) * 262144;
    convlstm_step<64, 32><<<256, 256, 0, stream>>>(h1out, h2in, h2out, c2, wh2, b2);

    fc_kernel<<<6, 1024, 0, stream>>>(h2out, fcw, fcb, out + t * 6);
  }

  // Final states: T=128 even -> final h lives in buf0 of each ping-pong.
  // d_out layout: pose[128*6] | h1[131072] | c1[131072] | h2[262144] | c2[262144]
  hipMemcpyAsync(out + 768,                   h1, 131072 * 4, hipMemcpyDeviceToDevice, stream);
  hipMemcpyAsync(out + 768 + 131072,          c1, 131072 * 4, hipMemcpyDeviceToDevice, stream);
  hipMemcpyAsync(out + 768 + 262144,          h2, 262144 * 4, hipMemcpyDeviceToDevice, stream);
  hipMemcpyAsync(out + 768 + 262144 + 262144, c2, 262144 * 4, hipMemcpyDeviceToDevice, stream);
}
